// TransformerBlock_8349416423926
// MI455X (gfx1250) — compile-verified, matrix operand order checked by reading, not run on hardware
//
#include <hip/hip_runtime.h>

// ---------------------------------------------------------------------------
// Transformer block for MI455X (gfx1250), compile-only WMMA pipeline.
//   B=4, S=2048, D=512, H=4  -> all GEMM dims divide the 128x256 block tile.
// All matmuls run on v_wmma_f32_16x16x32_bf16 (bf16 in, f32 accumulate).
// Global->LDS staging uses GLOBAL_LOAD_ASYNC_TO_LDS_B128 (ASYNCcnt) with a
// double-buffered K-loop pipeline; falls back to sync loads if unavailable.
// ---------------------------------------------------------------------------

#define B_ 4
#define S_ 2048
#define D_ 512
#define H_ 4
#define M_ (B_ * S_)   // 8192 rows (b*S + s)
#define F_ (4 * D_)    // 2048 FFN hidden
#define LDT 40         // padded LDS tile stride (elements) to break bank conflicts

typedef __attribute__((ext_vector_type(16))) __bf16 v16bf;
typedef __attribute__((ext_vector_type(8)))  float  v8f;
typedef __attribute__((ext_vector_type(4)))  int    vint4_t;  // builtin's pointee type

typedef __attribute__((address_space(1))) vint4_t* gvec4p;  // global int4 vec ptr
typedef __attribute__((address_space(3))) vint4_t* lvec4p;  // LDS int4 vec ptr

#if defined(__HIP_DEVICE_COMPILE__) &&                                   \
    __has_builtin(__builtin_amdgcn_global_load_async_to_lds_b128) &&     \
    __has_builtin(__builtin_amdgcn_s_wait_asynccnt)
#define ASYNC_LDS 1
#else
#define ASYNC_LDS 0
#endif

// Load a 16x32 bf16 fragment (ISA 7.12.2 16-bit A layout) from an LDS tile.
// lanes 0-15: M=lane, K = {0..7, 16..23}; lanes 16-31: M=lane-16, K = {8..15, 24..31}
// -> two 16-byte ds reads per lane.
static __device__ __forceinline__ v16bf frag_ld(const __bf16* t, int row0, int lane) {
  const int r  = row0 + (lane & 15);
  const int kb = (lane & 16) ? 8 : 0;
  const __bf16* p = t + r * LDT + kb;
  union { int4 q[2]; v16bf v; } u;
  u.q[0] = *(const int4*)(p);
  u.q[1] = *(const int4*)(p + 16);
  return u.v;
}

// Stage a ROWSx32 bf16 tile (row-major, leading dim `ld`) global -> LDS.
// 256 threads, 16B (8 x bf16) chunks, 4 chunks per row.
// Uses the CDNA5 async-to-LDS path (ASYNCcnt, no VGPR round trip) when available.
template<int ROWS>
static __device__ __forceinline__ void stage_tile(__bf16* dst, const __bf16* src,
                                                  int ld, int tid) {
#pragma unroll
  for (int i = 0; i < ROWS / 64; ++i) {
    const int c  = tid + i * 256;
    const int r  = c >> 2;
    const int kc = (c & 3) << 3;
#if ASYNC_LDS
    vint4_t* gsrc = (vint4_t*)(src + (size_t)r * ld + kc);  // drop const, retype
    vint4_t* ldst = (vint4_t*)(dst + r * LDT + kc);
    __builtin_amdgcn_global_load_async_to_lds_b128((gvec4p)gsrc, (lvec4p)ldst, 0, 0);
#else
    *(int4*)(dst + r * LDT + kc) = *(const int4*)(src + (size_t)r * ld + kc);
#endif
  }
}

static __device__ __forceinline__ void wait_stage() {
#if ASYNC_LDS
  __builtin_amdgcn_s_wait_asynccnt(0);
#endif
}

// One K=32 step for a 64x64 wave tile: 4 A-frags x 4 B-frags -> 16 WMMAs.
static __device__ __forceinline__ void mma_step(const __bf16* As, const __bf16* Bs,
                                                v8f acc[4][4], int wm, int wn, int lane) {
  v16bf a[4], b[4];
#pragma unroll
  for (int i = 0; i < 4; ++i) a[i] = frag_ld(As, wm + i * 16, lane);
#pragma unroll
  for (int j = 0; j < 4; ++j) b[j] = frag_ld(Bs, wn + j * 16, lane);
#pragma unroll
  for (int i = 0; i < 4; ++i)
#pragma unroll
    for (int j = 0; j < 4; ++j)
      acc[i][j] = __builtin_amdgcn_wmma_f32_16x16x32_bf16(
          false, a[i], false, b[j], (short)0, acc[i][j], false, false);
}

// Common prologue: 8 wave32s arranged 2(M) x 4(N) over a 128x256 block tile.
// Double-buffered LDS tiles: 2 * (128+256) * 40 * 2B = 60 KB (<320 KB/WGP).
#define GEMM_PRO()                                                  \
  __shared__ __bf16 As[2][128 * LDT];                               \
  __shared__ __bf16 Bs[2][256 * LDT];                               \
  const int tid  = (int)threadIdx.x;                                \
  const int lane = tid & 31;                                        \
  const int wv   = tid >> 5;                                        \
  const int wm   = (wv & 1) << 6;                                   \
  const int wn   = (wv >> 1) << 6;                                  \
  const int m0   = (int)blockIdx.y * 128;                           \
  const int n0   = (int)blockIdx.x * 256;                           \
  v8f acc[4][4];                                                    \
  {                                                                 \
    v8f z = {0.f, 0.f, 0.f, 0.f, 0.f, 0.f, 0.f, 0.f};              \
    for (int i = 0; i < 4; ++i)                                     \
      for (int j = 0; j < 4; ++j) acc[i][j] = z;                    \
  }

// Software-pipelined K loop: async-stage tile k+1 into buffer pp^1 while the
// matrix pipe consumes buffer pp; then wait ASYNCcnt==0, barrier, swap.
#define GEMM_K_LOOP(APTR, ALD, BPTR, BLD, KTOT)                                   \
  {                                                                               \
    int pp = 0;                                                                   \
    stage_tile<128>(As[0], (APTR) + (size_t)m0 * (ALD), (ALD), tid);              \
    stage_tile<256>(Bs[0], (BPTR) + (size_t)n0 * (BLD), (BLD), tid);              \
    wait_stage();                                                                 \
    __syncthreads();                                                              \
    for (int k0 = 32; k0 < (KTOT); k0 += 32) {                                    \
      stage_tile<128>(As[pp ^ 1], (APTR) + (size_t)m0 * (ALD) + k0, (ALD), tid);  \
      stage_tile<256>(Bs[pp ^ 1], (BPTR) + (size_t)n0 * (BLD) + k0, (BLD), tid);  \
      mma_step(As[pp], Bs[pp], acc, wm, wn, lane);                                \
      wait_stage();                                                               \
      __syncthreads();                                                            \
      pp ^= 1;                                                                    \
    }                                                                             \
    mma_step(As[pp], Bs[pp], acc, wm, wn, lane);                                  \
    __syncthreads();                                                              \
  }

// C/D layout (ISA 7.12.2): VGPR r, lanes0-15 -> M=r, N=lane; lanes16-31 -> M=8+r.
#define GEMM_EPI_BEGIN()                                                     \
  const int cn = lane & 15;                                                  \
  const int rh = (lane >> 4) << 3;                                           \
  _Pragma("unroll") for (int i = 0; i < 4; ++i)                              \
  _Pragma("unroll") for (int j = 0; j < 4; ++j)                              \
  _Pragma("unroll") for (int r = 0; r < 8; ++r) {                            \
    const int row = m0 + wm + i * 16 + rh + r;                               \
    const int col = n0 + wn + j * 16 + cn;                                   \
    const float val = acc[i][j][r];

#define GEMM_EPI_END() }

// ---------------------------------------------------------------------------
// Kernels
// ---------------------------------------------------------------------------

// Q/K/V projection: X[8192,512] x W_hT -> per-head bf16 activations.
// TRANS=false: out[h][m][e] row-major (Q, K). TRANS=true: out[h][b][e][s] (V^T).
template<bool TRANS>
__global__ __launch_bounds__(256) void k_qkv(const __bf16* __restrict__ X,
                                             const __bf16* __restrict__ WT,
                                             const float* __restrict__ bias,
                                             __bf16* __restrict__ out) {
  const int h = (int)blockIdx.z;
  GEMM_PRO();
  const __bf16* Wh = WT + (size_t)h * D_ * D_;
  GEMM_K_LOOP(X, D_, Wh, D_, D_);
  GEMM_EPI_BEGIN()
    const float v = val + bias[h * D_ + col];
    if (TRANS) {
      const int bb = row >> 11;          // row / S
      const int ss = row & (S_ - 1);     // row % S
      out[(((size_t)h * B_ + bb) * D_ + col) * S_ + ss] = (__bf16)v;
    } else {
      out[((size_t)h * M_ + row) * D_ + col] = (__bf16)v;
    }
  GEMM_EPI_END()
}

// scores[b,h] = scale * Q[b,h] K[b,h]^T  (B operand of QK^T is K row-major).
__global__ __launch_bounds__(256) void k_scores(const __bf16* __restrict__ Q,
                                                const __bf16* __restrict__ Km,
                                                float* __restrict__ Sc) {
  const int z = (int)blockIdx.z;  // b*H + h
  const int b = z >> 2;
  const int h = z & 3;
  GEMM_PRO();
  const __bf16* Ab = Q  + ((size_t)h * M_ + (size_t)b * S_) * D_;
  const __bf16* Bb = Km + ((size_t)h * M_ + (size_t)b * S_) * D_;
  GEMM_K_LOOP(Ab, D_, Bb, D_, D_);
  const float scale = 0.04419417382415922f;  // 1/sqrt(512)
  GEMM_EPI_BEGIN()
    Sc[((size_t)z * S_ + row) * S_ + col] = val * scale;
  GEMM_EPI_END()
}

// Row softmax over S=2048: one 256-thread block per row, fp32 math, bf16 out.
__global__ __launch_bounds__(256) void k_softmax(const float* __restrict__ Sc,
                                                 __bf16* __restrict__ P) {
  __shared__ float red[256];
  const int    tid = (int)threadIdx.x;
  const size_t row = (size_t)blockIdx.x;
  const float* rp  = Sc + row * S_;

  float v[8];
  float mx = -3.4e38f;
#pragma unroll
  for (int i = 0; i < 8; ++i) {
    v[i] = rp[tid + i * 256];
    mx   = fmaxf(mx, v[i]);
  }
  red[tid] = mx;
  __syncthreads();
  for (int s = 128; s > 0; s >>= 1) {
    if (tid < s) red[tid] = fmaxf(red[tid], red[tid + s]);
    __syncthreads();
  }
  mx = red[0];
  __syncthreads();

  float sum = 0.f;
#pragma unroll
  for (int i = 0; i < 8; ++i) {
    v[i] = __expf(v[i] - mx);
    sum += v[i];
  }
  red[tid] = sum;
  __syncthreads();
  for (int s = 128; s > 0; s >>= 1) {
    if (tid < s) red[tid] += red[tid + s];
    __syncthreads();
  }
  const float inv = 1.f / red[0];

  __bf16* op = P + row * S_;
#pragma unroll
  for (int i = 0; i < 8; ++i) op[tid + i * 256] = (__bf16)(v[i] * inv);
}

// ctx + head-mean fused: att[b] = (1/H) * sum_h P[b,h] @ V[b,h]  (V pre-transposed).
__global__ __launch_bounds__(256) void k_ctx(const __bf16* __restrict__ P,
                                             const __bf16* __restrict__ Vt,
                                             float* __restrict__ att,
                                             __bf16* __restrict__ attb) {
  const int b = (int)blockIdx.z;
  GEMM_PRO();
  for (int h = 0; h < H_; ++h) {
    const __bf16* Ab = P  + (size_t)(b * H_ + h) * S_ * S_;
    const __bf16* Bb = Vt + (size_t)(h * B_ + b) * (size_t)D_ * S_;
    GEMM_K_LOOP(Ab, S_, Bb, S_, S_);
  }
  const float inv = 1.0f / (float)H_;
  GEMM_EPI_BEGIN()
    const float  v = val * inv;
    const size_t o = ((size_t)b * S_ + row) * D_ + col;
    att[o]  = v;
    attb[o] = (__bf16)v;
  GEMM_EPI_END()
}

// FFN layer 1: hid = relu(att @ W1 + b1) -> bf16 [8192, 2048]
__global__ __launch_bounds__(256) void k_ffn1(const __bf16* __restrict__ A,
                                              const __bf16* __restrict__ W1T,
                                              const float* __restrict__ b1,
                                              __bf16* __restrict__ hid) {
  GEMM_PRO();
  GEMM_K_LOOP(A, D_, W1T, D_, D_);
  GEMM_EPI_BEGIN()
    float v = val + b1[col];
    v = v > 0.f ? v : 0.f;
    hid[(size_t)row * F_ + col] = (__bf16)v;
  GEMM_EPI_END()
}

// FFN layer 2 + residual: out = att + hid @ W2 + b2 -> f32 d_out
__global__ __launch_bounds__(256) void k_ffn2(const __bf16* __restrict__ hidb,
                                              const __bf16* __restrict__ W2T,
                                              const float* __restrict__ b2,
                                              const float* __restrict__ att,
                                              float* __restrict__ out) {
  GEMM_PRO();
  GEMM_K_LOOP(hidb, F_, W2T, F_, F_);
  GEMM_EPI_BEGIN()
    const size_t o = (size_t)row * D_ + col;
    out[o] = val + b2[col] + att[o];
  GEMM_EPI_END()
}

// fp32 -> bf16 elementwise
__global__ void k_cvt(const float* __restrict__ in, __bf16* __restrict__ outp, int n) {
  const int i = (int)blockIdx.x * 256 + (int)threadIdx.x;
  if (i < n) outp[i] = (__bf16)in[i];
}

// fp32 [R,C] -> bf16 [C,R] (batched over blockIdx.z) for B^T weight layout
__global__ void k_cvtT(const float* __restrict__ in, __bf16* __restrict__ outp,
                       int R, int C) {
  const size_t base = (size_t)blockIdx.z * (size_t)R * C;
  const int    i    = (int)blockIdx.x * 256 + (int)threadIdx.x;
  if (i < R * C) {
    const int r = i / C;
    const int c = i % C;
    outp[base + (size_t)c * R + r] = (__bf16)in[base + i];
  }
}

// ---------------------------------------------------------------------------
// Host launcher
// ---------------------------------------------------------------------------
extern "C" void kernel_launch(void* const* d_in, const int* in_sizes, int n_in,
                              void* d_out, int out_size, void* d_ws, size_t ws_size,
                              hipStream_t stream) {
  (void)in_sizes; (void)n_in; (void)out_size; (void)ws_size;

  const float* fencs = (const float*)d_in[0];
  const float* Wq    = (const float*)d_in[1];
  const float* bq    = (const float*)d_in[2];
  const float* Wk    = (const float*)d_in[3];
  const float* bk    = (const float*)d_in[4];
  const float* Wv    = (const float*)d_in[5];
  const float* bv    = (const float*)d_in[6];
  const float* W1    = (const float*)d_in[7];
  const float* b1    = (const float*)d_in[8];
  const float* W2    = (const float*)d_in[9];
  const float* b2    = (const float*)d_in[10];
  float*       out   = (float*)d_out;

  char*  wsp = (char*)d_ws;
  size_t off = 0;
  auto alloc = [&](size_t bytes) -> void* {
    off = (off + 255) & ~(size_t)255;
    void* p = wsp + off;
    off += bytes;
    return p;
  };

  __bf16* Xb   = (__bf16*)alloc((size_t)M_ * D_ * 2);             // fencs bf16
  __bf16* WqT  = (__bf16*)alloc((size_t)H_ * D_ * D_ * 2);        // [h][e][d]
  __bf16* WkT  = (__bf16*)alloc((size_t)H_ * D_ * D_ * 2);
  __bf16* WvT  = (__bf16*)alloc((size_t)H_ * D_ * D_ * 2);
  __bf16* W1T  = (__bf16*)alloc((size_t)D_ * F_ * 2);             // [F][D]
  __bf16* W2T  = (__bf16*)alloc((size_t)D_ * F_ * 2);             // [D][F]
  __bf16* Qb   = (__bf16*)alloc((size_t)H_ * M_ * D_ * 2);        // [h][m][e]
  __bf16* Kb   = (__bf16*)alloc((size_t)H_ * M_ * D_ * 2);        // [h][m][e]
  __bf16* Vt   = (__bf16*)alloc((size_t)H_ * M_ * D_ * 2);        // [h][b][e][s]
  float*  Sc   = (float*) alloc((size_t)B_ * H_ * S_ * S_ * 4);   // scores f32
  __bf16* Pb   = (__bf16*)alloc((size_t)B_ * H_ * S_ * S_ * 2);   // probs bf16
  float*  att  = (float*) alloc((size_t)M_ * D_ * 4);
  __bf16* attb = (__bf16*)alloc((size_t)M_ * D_ * 2);
  __bf16* hid  = (__bf16*)alloc((size_t)M_ * F_ * 2);

  // --- one-time precision / layout conversion ---
  {
    const int n = M_ * D_;
    k_cvt<<<dim3((n + 255) / 256), 256, 0, stream>>>(fencs, Xb, n);
  }
  k_cvtT<<<dim3((D_ * D_ + 255) / 256, 1, H_), 256, 0, stream>>>(Wq, WqT, D_, D_);
  k_cvtT<<<dim3((D_ * D_ + 255) / 256, 1, H_), 256, 0, stream>>>(Wk, WkT, D_, D_);
  k_cvtT<<<dim3((D_ * D_ + 255) / 256, 1, H_), 256, 0, stream>>>(Wv, WvT, D_, D_);
  k_cvtT<<<dim3((D_ * F_ + 255) / 256, 1, 1), 256, 0, stream>>>(W1, W1T, D_, F_);
  k_cvtT<<<dim3((D_ * F_ + 255) / 256, 1, 1), 256, 0, stream>>>(W2, W2T, F_, D_);

  // --- QKV projections (batched over heads in grid.z) ---
  k_qkv<false><<<dim3(D_ / 256, M_ / 128, H_), 256, 0, stream>>>(Xb, WqT, bq, Qb);
  k_qkv<false><<<dim3(D_ / 256, M_ / 128, H_), 256, 0, stream>>>(Xb, WkT, bk, Kb);
  k_qkv<true> <<<dim3(D_ / 256, M_ / 128, H_), 256, 0, stream>>>(Xb, WvT, bv, Vt);

  // --- attention ---
  k_scores <<<dim3(S_ / 256, S_ / 128, B_ * H_), 256, 0, stream>>>(Qb, Kb, Sc);
  k_softmax<<<dim3(B_ * H_ * S_), 256, 0, stream>>>(Sc, Pb);
  k_ctx    <<<dim3(D_ / 256, S_ / 128, B_), 256, 0, stream>>>(Pb, Vt, att, attb);

  // --- FFN + residual ---
  k_ffn1<<<dim3(F_ / 256, M_ / 128), 256, 0, stream>>>(attb, W1T, b1, hid);
  k_ffn2<<<dim3(D_ / 256, M_ / 128), 256, 0, stream>>>(hid, W2T, b2, att, out);
}
